// SunConditionedStochasticCloudModel_91242285236656
// MI455X (gfx1250) — matrix operand order repaired and checked by reading, not compiled, not run on hardware
//
#include <hip/hip_runtime.h>
#include <hip/hip_bf16.h>
#include <math.h>

// =====================================================================
// SunConditionedStochasticCloudModel forward for MI455X (gfx1250).
//
// Matrix path: v_wmma_f32_16x16x32_f16 implicit GEMM (wave32 WMMA),
// f32 activations in HBM, padded f16 weights (pre-converted per call),
// f32 accumulate.  Weight tiles staged into LDS by the Tensor Data
// Mover (tensor_load_to_lds + s_wait_tensorcnt) when available.
// im2col staging: batched branchless loads, 32-bit incremental offsets.
// =====================================================================

typedef __attribute__((ext_vector_type(16))) _Float16 v16h;
typedef __attribute__((ext_vector_type(2)))  _Float16 v2h;
typedef __attribute__((ext_vector_type(8)))  float    v8f;
typedef __attribute__((ext_vector_type(4)))  unsigned int v4u;
typedef __attribute__((ext_vector_type(8)))  int      v8i;
typedef __attribute__((ext_vector_type(4)))  int      v4i;

#if __has_builtin(__builtin_amdgcn_tensor_load_to_lds)
#define USE_TDM 1
#else
#define USE_TDM 0
#endif

#define ACT_NONE    0
#define ACT_GELU    1
#define ACT_SIGMOID 2
#define ACT_TANH    3

#define BN_INV 0.99999500003749969f   // 1/sqrt(1+1e-5)

__device__ __forceinline__ float gelu_f(float x) {
  return 0.5f * x * (1.0f + erff(x * 0.70710678118654752f));
}
__device__ __forceinline__ float sigmoid_f(float x) { return 1.0f / (1.0f + expf(-x)); }
__device__ __forceinline__ float softplus_f(float x) { return (x > 20.f) ? x : log1pf(expf(x)); }

__device__ __forceinline__ float act_apply(float v, int act) {
  switch (act) {
    case ACT_GELU:    return gelu_f(v);
    case ACT_SIGMOID: return sigmoid_f(v);
    case ACT_TANH:    return tanhf(v);
    default:          return v;
  }
}

// f32 -> f16 weight conversion with tile padding (zeros in pad region).
__global__ void wcvt_k(const float* __restrict__ src, _Float16* __restrict__ dst,
                       int M, int K, int Mp, int Kp) {
  long i = (long)blockIdx.x * blockDim.x + threadIdx.x;
  if (i >= (long)Mp * Kp) return;
  int m = (int)(i / Kp);
  int k = (int)(i - (long)m * Kp);
  float v = (m < M && k < K) ? src[(long)m * K + k] : 0.0f;
  dst[i] = (_Float16)v;
}

// ---------------------------------------------------------------------
// Implicit-GEMM conv.  GEMM view: M = Cout, N = Nimg*OH*OW,
// K = (Cin1+Cin2)*KH*KW (padded to Kp).  Split input along channels
// handles the ConvLSTM concat([xt, h]) without a copy.
// 256 threads = 8 waves; tile 64(M) x 128(N) x 32(K);
// wave = one 16-row M subtile x four 16-col N subtiles -> 4 WMMA/step.
// Weight tile (64x32 f16 = 4KB) staged via TDM when available.
// ---------------------------------------------------------------------
#define CBM 64
#define CBN 128
#define CBK 32

__global__ __launch_bounds__(256)
void conv2d_wmma(const float* __restrict__ X1, int Cin1, long xs1, long xo1,
                 const float* __restrict__ X2, int Cin2, long xs2, long xo2,
                 const _Float16* __restrict__ WtH, int Kp,
                 const float* __restrict__ scale, float scale_mul,
                 const float* __restrict__ bias,
                 const float* __restrict__ resid,
                 float* __restrict__ Y,
                 int Nimg, int H, int W, int OH, int OW,
                 int Cout, int KH, int KW, int stride, int pad, int act)
{
  const int Cin = Cin1 + Cin2;
  const int K   = Cin * KH * KW;
  const int KHW = KH * KW;
  const int HW  = H * W;
  const int OHW = OH * OW;
  const long Ngemm = (long)Nimg * OHW;

  __shared__ __align__(64) _Float16 As[CBM][CBK];   // [m][k]
  __shared__ __align__(64) _Float16 Bs[CBN][CBK];   // [n][k]

  const int tid  = threadIdx.x;
  const int lane = tid & 31;
  const int wid  = tid >> 5;
  const long m_blk = (long)blockIdx.y * CBM;
  const long n_blk = (long)blockIdx.x * CBN;

  const int mfrag = (wid & 3) * 16;   // wave's 16-row M subtile
  const int nbase = (wid >> 2) * 64;  // wave's 64-col N strip (4 subtiles)

  v8f acc[4] = {};

  // staging coordinates
  const int a_m  = tid >> 2;          // A fallback: 8 halves per thread
  const int a_k0 = (tid & 3) * 8;
  const int b_n  = tid >> 1;          // B: 16 consecutive k per thread
  const int b_k0 = (tid & 1) * 16;

  // decompose this thread's im2col output pixel once
  const long gn = n_blk + b_n;
  int nb = 0, oy = 0, ox = 0;
  bool n_ok = false;
  if (gn < Ngemm) {
    nb = (int)(gn / OHW);
    int r = (int)(gn - (long)nb * OHW);
    oy = r / OW; ox = r - oy * OW;
    n_ok = true;
  }
  const int iy0 = oy * stride - pad;
  const int ix0 = ox * stride - pad;
  const int offBase = iy0 * W + ix0;
  const int Cin1HW  = Cin1 * HW;
  const float* base1 = X1 + (long)nb * xs1 + xo1;           // 64-bit math hoisted
  const float* base2 = X2 ? (X2 + (long)nb * xs2 + xo2) : base1;

#if USE_TDM
  const unsigned lds_As_base = (unsigned)(size_t)&As[0][0];  // LDS byte address (low 32 bits)
  const int Mp = (int)gridDim.y * CBM;
#endif

  for (int k0 = 0; k0 < Kp; k0 += CBK) {
    // ---- stage A tile (padded f16 weights, guard-free) ----
#if USE_TDM
    if (wid == 0) {
      const unsigned long long ga =
          (unsigned long long)(size_t)(WtH + (size_t)m_blk * Kp + k0);
      v4u g0;
      g0[0] = 1u;                                   // count = 1 descriptor
      g0[1] = lds_As_base;                          // lds_addr (bytes)
      g0[2] = (unsigned)(ga & 0xffffffffull);       // global_addr lo
      g0[3] = (unsigned)((ga >> 32) & 0x1ffffffull) | 0x80000000u; // hi | type=2
      v8i g1;
      g1[0] = 0x00010000;                           // data_size = 2B
      g1[1] = (int)(((unsigned)Kp & 0xffffu) << 16);               // tensor_dim0 lo
      g1[2] = (int)((((unsigned)Kp >> 16) & 0xffffu) |
                    (((unsigned)Mp & 0xffffu) << 16));             // dim0 hi | dim1 lo
      g1[3] = (int)((((unsigned)Mp >> 16) & 0xffffu) | (32u << 16)); // tile_dim0=32
      g1[4] = 64;                                   // tile_dim1=64, tile_dim2=0
      g1[5] = Kp;                                   // tensor_dim0_stride lo32
      g1[6] = 0;
      g1[7] = 0;
      v4i gz = {0, 0, 0, 0};
#if __clang_major__ >= 23
      v8i gz8 = {0, 0, 0, 0, 0, 0, 0, 0};
      __builtin_amdgcn_tensor_load_to_lds(g0, g1, gz, gz, gz8, 0);
#else
      __builtin_amdgcn_tensor_load_to_lds(g0, g1, gz, gz, 0);
#endif
#if __has_builtin(__builtin_amdgcn_s_wait_tensorcnt)
      __builtin_amdgcn_s_wait_tensorcnt((short)0);
#else
      asm volatile("s_wait_tensorcnt 0x0" ::: "memory");
#endif
    }
#else
    {
      const _Float16* src = WtH + (size_t)(m_blk + a_m) * Kp + k0 + a_k0;
      __builtin_prefetch(src + CBK, 0, 0);          // global_prefetch_b8
      *(uint4*)&As[a_m][a_k0] = *(const uint4*)src;
    }
#endif

    // ---- stage B tile: branchless im2col, batched loads ----
    {
      const int gk0 = k0 + b_k0;
      int ci = gk0 / KHW;
      int rr = gk0 - ci * KHW;
      int kh = rr / KW;
      int kw = rr - kh * KW;
      int roff = ci * HW + kh * W + kw;   // 32-bit running offset
      float vv[16];
      bool  okb[16];
#pragma unroll
      for (int i = 0; i < 16; ++i) {
        const int gk = gk0 + i;
        const int iy = iy0 + kh;
        const int ix = ix0 + kw;
        const bool in1 = (ci < Cin1);
        const bool ok = n_ok && (gk < K) &&
                        ((unsigned)iy < (unsigned)H) && ((unsigned)ix < (unsigned)W);
        okb[i] = ok;
        const float* Xp = in1 ? base1 : base2;
        int off = in1 ? (roff + offBase) : (roff + offBase - Cin1HW);
        off = ok ? off : 0;
        vv[i] = Xp[off];                  // loads batch; one wait later
        // advance (kw,kh,ci,roff) without divides
        kw += 1; roff += 1;
        const bool cw = (kw == KW);
        kw = cw ? 0 : kw; kh += cw ? 1 : 0; roff += cw ? (W - KW) : 0;
        const bool chh = (kh == KH);
        kh = chh ? 0 : kh; ci += chh ? 1 : 0; roff += chh ? (HW - KH * W) : 0;
      }
#pragma unroll
      for (int i = 0; i < 16; ++i)
        Bs[b_n][b_k0 + i] = (_Float16)(okb[i] ? vv[i] : 0.0f);
    }
    __syncthreads();

    // ---- fragments (contiguous k = 16*(lane/16) + s, same order in A and B) ----
    v16h afrag;
    {
      const int row = mfrag + (lane & 15);
      const int kb  = (lane >= 16) ? 16 : 0;
#pragma unroll
      for (int j = 0; j < 8; ++j) {
        const int kk = kb + 2 * j;
        v2h p = *(const v2h*)&As[row][kk];
        afrag[2 * j]     = p[0];
        afrag[2 * j + 1] = p[1];
      }
    }
    const int kb2 = (lane >= 16) ? 16 : 0;
#pragma unroll
    for (int q = 0; q < 4; ++q) {
      v16h bf = *(const v16h*)&Bs[nbase + q * 16 + (lane & 15)][kb2];
      acc[q] = __builtin_amdgcn_wmma_f32_16x16x32_f16(false, afrag, false, bf,
                                                      (short)0, acc[q], false, false);
    }
    __syncthreads();
  }

  // ---- epilogue: D rows r -> M = r (+8 for lanes 16..31) ----
#pragma unroll
  for (int q = 0; q < 4; ++q) {
    const long gncol = n_blk + nbase + q * 16 + (lane & 15);
    if (gncol >= Ngemm) continue;
    const int nb2 = (int)(gncol / OHW);
    const int pix = (int)(gncol - (long)nb2 * OHW);
#pragma unroll
    for (int r = 0; r < 8; ++r) {
      const long gm = m_blk + mfrag + r + ((lane < 16) ? 0 : 8);
      if (gm >= Cout) continue;
      float v = acc[q][r];
      const float s = scale ? (scale[gm] * scale_mul) : 1.0f;
      const float b = bias ? bias[gm] : 0.0f;
      v = v * s + b;
      const long oidx = (long)nb2 * Cout * OHW + gm * OHW + pix;
      if (resid) v += resid[oidx];
      Y[oidx] = act_apply(v, act);
    }
  }
}

// ---------------------------------------------------------------------
// Small GEMM for lin(): C[M<=8, N] = X[M,K] * W[N,K]^T + b.
// Padded f16 weights, batched branchless activation staging.
// ---------------------------------------------------------------------
#define GBN 128
#define GBK 32

__global__ __launch_bounds__(256)
void gemm8_wmma(const float* __restrict__ X1, int K1,
                const float* __restrict__ X2, int K2,
                const _Float16* __restrict__ WtH, int Kp,
                const float* __restrict__ bias,
                float* __restrict__ Y, long y_row_stride, long y_off,
                int Nrows, int N, int act, float post_scale)
{
  const int K = K1 + K2;
  __shared__ __align__(64) _Float16 As[16][GBK];
  __shared__ __align__(64) _Float16 Bs[GBN][GBK];

  const int tid  = threadIdx.x;
  const int lane = tid & 31;
  const int wid  = tid >> 5;
  const int n_blk = blockIdx.x * GBN;
  const int nfrag = wid * 16;

  v8f acc = {};

  for (int k0 = 0; k0 < Kp; k0 += GBK) {
    // stage A (activations): batched branchless, 2 per thread
    float av[2];
    bool  aok[2];
#pragma unroll
    for (int i = 0; i < 2; ++i) {
      const int e = tid * 2 + i;
      const int m = e >> 5, k = e & 31;
      const int gk = k0 + k;
      const bool ok = (m < Nrows) && (gk < K);
      const int mc  = ok ? m : 0;
      const int gkc = ok ? gk : 0;
      const bool in1 = (gkc < K1);
      const float* Xp = in1 ? X1 : (X2 ? X2 : X1);
      const int off = in1 ? (mc * K1 + gkc) : (mc * K2 + (gkc - K1));
      av[i]  = Xp[ok ? off : 0];
      aok[i] = ok;
    }
#pragma unroll
    for (int i = 0; i < 2; ++i) {
      const int e = tid * 2 + i;
      As[e >> 5][e & 31] = (_Float16)(aok[i] ? av[i] : 0.0f);
    }
    // stage B (padded f16 weights): guard-free 2x16B per thread
    {
      const int n   = tid >> 1;
      const int kk0 = (tid & 1) * 16;
      const _Float16* src = WtH + (size_t)(n_blk + n) * Kp + k0 + kk0;
      __builtin_prefetch(src + GBK, 0, 0);
      *(uint4*)&Bs[n][kk0]     = *(const uint4*)(src);
      *(uint4*)&Bs[n][kk0 + 8] = *(const uint4*)(src + 8);
    }
    __syncthreads();

    v16h afrag;
    {
      const int row = lane & 15;
      const int kb  = (lane >= 16) ? 16 : 0;
#pragma unroll
      for (int j = 0; j < 8; ++j) {
        const int kk = kb + 2 * j;
        v2h p = *(const v2h*)&As[row][kk];
        afrag[2 * j]     = p[0];
        afrag[2 * j + 1] = p[1];
      }
    }
    const int kb2 = (lane >= 16) ? 16 : 0;
    v16h bfrag = *(const v16h*)&Bs[nfrag + (lane & 15)][kb2];

    acc = __builtin_amdgcn_wmma_f32_16x16x32_f16(false, afrag, false, bfrag,
                                                 (short)0, acc, false, false);
    __syncthreads();
  }

  const int gn = n_blk + nfrag + (lane & 15);
  if (gn < N) {
    const float b = bias ? bias[gn] : 0.0f;
#pragma unroll
    for (int r = 0; r < 8; ++r) {
      const int m = r + ((lane < 16) ? 0 : 8);
      if (m < Nrows) {
        float v = act_apply(acc[r] + b, act) * post_scale;
        Y[(long)m * y_row_stride + y_off + gn] = v;
      }
    }
  }
}

// ------------------------- pointwise kernels -------------------------

__global__ void fill_k(float* __restrict__ p, float v, long n) {
  long i = (long)blockIdx.x * blockDim.x + threadIdx.x;
  if (i < n) p[i] = v;
}

__global__ void copy_mat_k(float* __restrict__ dst, long dstride,
                           const float* __restrict__ src, long sstride,
                           int rows, int cols) {
  int i = blockIdx.x * blockDim.x + threadIdx.x;
  if (i >= rows * cols) return;
  int r = i / cols, c = i - r * cols;
  dst[(long)r * dstride + c] = src[(long)r * sstride + c];
}

__global__ void lstm_cell_k(const float* __restrict__ G,
                            float* __restrict__ C, float* __restrict__ H,
                            int total) {
  int idx = blockIdx.x * blockDim.x + threadIdx.x;
  if (idx >= total) return;
  const int CHW = 256 * 1024;
  const int b = idx / CHW;
  const int r = idx - b * CHW;
  const long gbase = (long)b * 4 * CHW;
  const float gi = G[gbase + r];
  const float gf = G[gbase + CHW + r];
  const float go = G[gbase + 2 * CHW + r];
  const float gg = G[gbase + 3 * CHW + r];
  float c = sigmoid_f(gf) * C[idx] + sigmoid_f(gi) * tanhf(gg);
  C[idx] = c;
  H[idx] = sigmoid_f(go) * tanhf(c);
}

__global__ void gru_cell_k(const float* __restrict__ gi,
                           const float* __restrict__ gh,
                           float* __restrict__ h, int D, int total) {
  int idx = blockIdx.x * blockDim.x + threadIdx.x;
  if (idx >= total) return;
  const int b = idx / D, j = idx - b * D;
  const float* gib = gi + (long)b * 3 * D;
  const float* ghb = gh + (long)b * 3 * D;
  const float r  = sigmoid_f(gib[j] + ghb[j]);
  const float zz = sigmoid_f(gib[D + j] + ghb[D + j]);
  const float n  = tanhf(gib[2 * D + j] + r * ghb[2 * D + j]);
  h[idx] = (1.f - zz) * n + zz * h[idx];
}

__global__ void mean_pool_k(const float* __restrict__ X, float* __restrict__ Y, int HW) {
  __shared__ float sm[256];
  const int bc = blockIdx.x;
  const float* p = X + (long)bc * HW;
  float s = 0.f;
  for (int i = threadIdx.x; i < HW; i += 256) s += p[i];
  sm[threadIdx.x] = s;
  __syncthreads();
  for (int off = 128; off > 0; off >>= 1) {
    if (threadIdx.x < off) sm[threadIdx.x] += sm[threadIdx.x + off];
    __syncthreads();
  }
  if (threadIdx.x == 0) Y[bc] = sm[0] / (float)HW;
}

__global__ void latent_finalize_k(const float* __restrict__ sm_, const float* __restrict__ so_,
                                  const float* __restrict__ sg_, const float* __restrict__ ss_,
                                  float* __restrict__ z, float* __restrict__ kl) {
  const int b = threadIdx.x >> 5;
  const int lane = threadIdx.x & 31;
  const float* S[4] = { sm_, so_, sg_, ss_ };
  const int d[4]   = { 64, 32, 32, 16 };
  const int off[4] = { 0, 64, 96, 128 };
  float acc = 0.f;
  for (int hI = 0; hI < 4; ++hI) {
    const float* st = S[hI] + (long)b * 2 * d[hI];
    for (int j = lane; j < d[hI]; j += 32) {
      const float mu = st[j];
      float lv = st[d[hI] + j];
      lv = fminf(fmaxf(lv, -6.f), 4.f);
      acc += 0.5f * (expf(lv) + mu * mu - 1.f - lv);
      z[b * 144 + off[hI] + j] = mu;
    }
  }
  for (int o = 16; o > 0; o >>= 1) acc += __shfl_down(acc, o, 32);
  if (lane == 0) kl[b] = acc;
}

__global__ void transm_k(const float* __restrict__ opac, const float* __restrict__ gap,
                         float* __restrict__ tr, int total) {
  int i = blockIdx.x * blockDim.x + threadIdx.x;
  if (i >= total) return;
  float t = (1.f - opac[i]) * (0.25f + 0.75f * gap[i]);
  tr[i] = fminf(fmaxf(t, 0.f), 1.f);
}

__global__ void sun_attn_k(const float* __restrict__ sun_xy,
                           const float* __restrict__ trans_t, const float* __restrict__ gap_t,
                           const float* __restrict__ opac_t,
                           float* __restrict__ s_tr, float* __restrict__ s_gap,
                           float* __restrict__ s_op) {
  const int b = blockIdx.x;
  const float sx = sun_xy[b * 2 + 0] * 0.125f;
  const float sy = sun_xy[b * 2 + 1] * 0.125f;
  __shared__ float red[4][256];
  float w = 0.f, t = 0.f, g = 0.f, o = 0.f;
  const long base = (long)b * 15 * 1024 + 14 * 1024;
  for (int p = threadIdx.x; p < 1024; p += 256) {
    const int y = p >> 5, x = p & 31;
    const float dx = (float)x - sx, dy = (float)y - sy;
    const float a = expf(-(dx * dx + dy * dy) * (1.0f / 12.5f));
    w += a;
    t += a * trans_t[base + p];
    g += a * gap_t[base + p];
    o += a * opac_t[base + p];
  }
  red[0][threadIdx.x] = w; red[1][threadIdx.x] = t;
  red[2][threadIdx.x] = g; red[3][threadIdx.x] = o;
  __syncthreads();
  for (int s2 = 128; s2 > 0; s2 >>= 1) {
    if (threadIdx.x < s2)
      for (int q = 0; q < 4; ++q) red[q][threadIdx.x] += red[q][threadIdx.x + s2];
    __syncthreads();
  }
  if (threadIdx.x == 0) {
    const float den = fmaxf(red[0][0], 1e-6f);
    s_tr[b] = red[1][0] / den;
    s_gap[b] = red[2][0] / den;
    s_op[b] = red[3][0] / den;
  }
}

__global__ void qin_build_k(const float* __restrict__ pooled, const float* __restrict__ pv_feat,
                            const float* __restrict__ solar,
                            const float* __restrict__ s_tr, const float* __restrict__ s_gap,
                            const float* __restrict__ s_op, const float* __restrict__ sun_t,
                            float* __restrict__ qin) {
  int i = blockIdx.x * blockDim.x + threadIdx.x;
  if (i >= 8 * 329) return;
  const int b = i / 329, j = i - b * 329;
  float v;
  if (j < 256)      v = pooled[b * 256 + j];
  else if (j < 320) v = pv_feat[b * 64 + (j - 256)];
  else if (j < 325) v = solar[b * 5 + (j - 320)];
  else if (j == 325) v = s_tr[b];
  else if (j == 326) v = s_gap[b];
  else if (j == 327) v = s_op[b];
  else               v = sun_t[b * 15 + 14];
  qin[i] = v;
}

__global__ void quantile_k(const float* __restrict__ qc, const float* __restrict__ ql,
                           const float* __restrict__ qu, float* __restrict__ out) {
  const int b = threadIdx.x;
  if (b >= 8) return;
  const float q50 = qc[b];
  out[b * 3 + 0] = q50 - softplus_f(ql[b]);
  out[b * 3 + 1] = q50;
  out[b * 3 + 2] = q50 + softplus_f(qu[b]);
}

// ---------------------------------------------------------------------
// Parameter index map (jax pytree flatten = alphabetical dict keys).
// ---------------------------------------------------------------------
enum {
  IN_IMAGES = 0, IN_PV = 1, IN_SOLAR = 2, IN_SUNXY = 3,
  P_ENC_B4 = 4, P_ENC_BN1_B, P_ENC_BN1_G, P_ENC_BN2_B, P_ENC_BN2_G,
  P_ENC_BN3_B, P_ENC_BN3_G,
  P_RB1_BN1_B, P_RB1_BN1_G, P_RB1_BN2_B, P_RB1_BN2_G, P_RB1_W1, P_RB1_W2,
  P_RB2_BN1_B, P_RB2_BN1_G, P_RB2_BN2_B, P_RB2_BN2_G, P_RB2_W1, P_RB2_W2,
  P_RB3_BN1_B, P_RB3_BN1_G, P_RB3_BN2_B, P_RB3_BN2_G, P_RB3_W1, P_RB3_W2,
  P_ENC_W1, P_ENC_W2, P_ENC_W3, P_ENC_W4,
  P_GAPDEC_L1_B, P_GAPDEC_L1_W, P_GAPDEC_L2_B, P_GAPDEC_L2_W,
  P_GRU_BHH, P_GRU_BIH, P_GRU_WHH, P_GRU_WIH,
  P_HID_B, P_HID_W,
  P_LAT_GAP_L1_B, P_LAT_GAP_L1_W, P_LAT_GAP_L2_B, P_LAT_GAP_L2_W,
  P_LAT_MOT_L1_B, P_LAT_MOT_L1_W, P_LAT_MOT_L2_B, P_LAT_MOT_L2_W,
  P_LAT_OPA_L1_B, P_LAT_OPA_L1_W, P_LAT_OPA_L2_B, P_LAT_OPA_L2_W,
  P_LAT_SUN_L1_B, P_LAT_SUN_L1_W, P_LAT_SUN_L2_B, P_LAT_SUN_L2_W,
  P_LSTM_B, P_LSTM_W,
  P_MOTDEC_L1_B, P_MOTDEC_L1_W, P_MOTDEC_L2_B, P_MOTDEC_L2_W,
  P_OPADEC_L1_B, P_OPADEC_L1_W, P_OPADEC_L2_B, P_OPADEC_L2_W,
  P_PV_L1_B, P_PV_L1_W, P_PV_L2_B, P_PV_L2_W,
  P_Q_CENTER_B, P_Q_CENTER_W, P_Q_L1_B, P_Q_L1_W, P_Q_L2_B, P_Q_L2_W,
  P_Q_LOWER_B, P_Q_LOWER_W, P_Q_UPPER_B, P_Q_UPPER_W,
  P_REC_B1, P_REC_B2, P_REC_W1, P_REC_W2,
  P_REF_BN1_B, P_REF_BN1_G, P_REF_BN2_B, P_REF_BN2_G, P_REF_W1, P_REF_W2,
  P_SUNDEC_L1_B, P_SUNDEC_L1_W, P_SUNDEC_L2_B, P_SUNDEC_L2_W,
  P_TRANS_L1_B, P_TRANS_L1_W, P_TRANS_L2_B, P_TRANS_L2_W
};

// -------------------------- host-side helpers -------------------------

static inline long rnd(long x, long m) { return (x + m - 1) / m * m; }

static void launch_conv(hipStream_t s,
                        const float* X1, int Cin1, long xs1, long xo1,
                        const float* X2, int Cin2, long xs2, long xo2,
                        const _Float16* WtH, const float* scale, float scale_mul,
                        const float* bias, const float* resid,
                        float* Y, int Nimg, int H, int W, int stride, int pad,
                        int Cout, int KH, int KW, int act) {
  const int OH = (H + 2 * pad - KH) / stride + 1;
  const int OW = (W + 2 * pad - KW) / stride + 1;
  const long Ngemm = (long)Nimg * OH * OW;
  const int Kp = (int)rnd((long)(Cin1 + Cin2) * KH * KW, 32);
  dim3 grid((unsigned)((Ngemm + CBN - 1) / CBN), (unsigned)((Cout + CBM - 1) / CBM));
  conv2d_wmma<<<grid, 256, 0, s>>>(X1, Cin1, xs1, xo1, X2, Cin2, xs2, xo2,
                                   WtH, Kp, scale, scale_mul, bias, resid, Y,
                                   Nimg, H, W, OH, OW, Cout, KH, KW, stride, pad, act);
}

static void launch_gemm(hipStream_t s,
                        const float* X1, int K1, const float* X2, int K2,
                        const _Float16* WtH, const float* bias,
                        float* Y, long y_row_stride, long y_off,
                        int N, int act, float post_scale) {
  const int Kp = (int)rnd(K1 + K2, 32);
  dim3 grid((unsigned)((N + GBN - 1) / GBN));
  gemm8_wmma<<<grid, 256, 0, s>>>(X1, K1, X2, K2, WtH, Kp, bias, Y,
                                  y_row_stride, y_off, 8, N, act, post_scale);
}

// ============================ kernel_launch ===========================

extern "C" void kernel_launch(void* const* d_in, const int* in_sizes, int n_in,
                              void* d_out, int out_size, void* d_ws, size_t ws_size,
                              hipStream_t stream) {
  (void)in_sizes; (void)n_in; (void)out_size; (void)ws_size;
  auto F = [&](int i) -> const float* { return (const float*)d_in[i]; };
  float* ws = (float*)d_ws;

  // ---- f32 workspace layout ----
  const long PING = 64L * 32 * 128 * 128;     // 33,554,432
  float* bufA = ws;
  float* bufB = ws + PING;
  float* bufC = ws + 2 * PING;
  long o = 3 * PING;
  float* hbuf  = ws + o; o += 8L * 256 * 1024;
  float* cbuf  = ws + o; o += 8L * 256 * 1024;
  float* gates = ws + o; o += 8L * 1024 * 1024;
  float* tmp1  = ws + o; o += 8L * 256 * 1024;
  float* tmp2  = ws + o; o += 8L * 256 * 1024;
  float* pooled = ws + o; o += 8 * 256;
  float* st_mot = ws + o; o += 8 * 128;
  float* st_opa = ws + o; o += 8 * 64;
  float* st_gap = ws + o; o += 8 * 64;
  float* st_sun = ws + o; o += 8 * 32;
  float* zbuf  = ws + o; o += 8 * 144;
  float* hid   = ws + o; o += 8 * 128;
  float* czbuf = ws + o; o += 8 * 144;
  float* gibuf = ws + o; o += 8 * 384;
  float* ghbuf = ws + o; o += 8 * 384;
  float* t256  = ws + o; o += 8 * 256;
  float* tsbuf = ws + o; o += 8 * 288;
  float* t64   = ws + o; o += 8 * 64;
  float* pvt   = ws + o; o += 8 * 64;
  float* pvfeat = ws + o; o += 8 * 64;
  float* str_  = ws + o; o += 8;
  float* sgap_ = ws + o; o += 8;
  float* sop_  = ws + o; o += 8;
  float* qinb  = ws + o; o += 8 * 329;
  float* f1    = ws + o; o += 8 * 128;
  float* f2    = ws + o; o += 8 * 64;
  float* qc    = ws + o; o += 8;
  float* ql    = ws + o; o += 8;
  float* qu    = ws + o; o += 8;

  // ---- f16 weight region (padded, zero-filled pad) ----
  o = rnd(o, 8);                               // 32B alignment for uint4 copies
  _Float16* wh_base = (_Float16*)(ws + o);
  long wh_off = 0;
  auto cvt = [&](int idx, long M, long K, long Mp, long Kp) -> const _Float16* {
    _Float16* dst = wh_base + wh_off;
    const long tot = Mp * Kp;
    wcvt_k<<<(unsigned)((tot + 255) / 256), 256, 0, stream>>>(
        (const float*)d_in[idx], dst, (int)M, (int)K, (int)Mp, (int)Kp);
    wh_off += tot;
    return dst;
  };
  auto cvtc = [&](int idx, long M, long K) { return cvt(idx, M, K, rnd(M, 64), rnd(K, 32)); };
  auto cvtg = [&](int idx, long M, long K) { return cvt(idx, M, K, rnd(M, 128), rnd(K, 32)); };

  // conv weights (M=Cout, K=Cin*KH*KW)
  const _Float16* W_ENC1  = cvtc(P_ENC_W1, 32, 150);
  const _Float16* W_RB1A  = cvtc(P_RB1_W1, 32, 288);
  const _Float16* W_RB1B  = cvtc(P_RB1_W2, 32, 288);
  const _Float16* W_ENC2  = cvtc(P_ENC_W2, 64, 288);
  const _Float16* W_RB2A  = cvtc(P_RB2_W1, 64, 576);
  const _Float16* W_RB2B  = cvtc(P_RB2_W2, 64, 576);
  const _Float16* W_ENC3  = cvtc(P_ENC_W3, 96, 576);
  const _Float16* W_RB3A  = cvtc(P_RB3_W1, 96, 864);
  const _Float16* W_RB3B  = cvtc(P_RB3_W2, 96, 864);
  const _Float16* W_ENC4  = cvtc(P_ENC_W4, 128, 864);
  const _Float16* W_LSTM  = cvtc(P_LSTM_W, 1024, 3456);
  const _Float16* W_REF1  = cvtc(P_REF_W1, 256, 2304);
  const _Float16* W_REF2  = cvtc(P_REF_W2, 256, 2304);
  const _Float16* W_RECA  = cvtc(P_REC_W1, 128, 2304);
  const _Float16* W_RECB  = cvtc(P_REC_W2, 1, 128);
  // gemm weights (M=N_out, K=in)
  const _Float16* W_LM1 = cvtg(P_LAT_MOT_L1_W, 256, 256);
  const _Float16* W_LM2 = cvtg(P_LAT_MOT_L2_W, 128, 256);
  const _Float16* W_LO1 = cvtg(P_LAT_OPA_L1_W, 256, 256);
  const _Float16* W_LO2 = cvtg(P_LAT_OPA_L2_W, 64, 256);
  const _Float16* W_LG1 = cvtg(P_LAT_GAP_L1_W, 256, 256);
  const _Float16* W_LG2 = cvtg(P_LAT_GAP_L2_W, 64, 256);
  const _Float16* W_LS1 = cvtg(P_LAT_SUN_L1_W, 256, 256);
  const _Float16* W_LS2 = cvtg(P_LAT_SUN_L2_W, 32, 256);
  const _Float16* W_HID = cvtg(P_HID_W, 128, 144);
  const _Float16* W_WIH = cvtg(P_GRU_WIH, 384, 144);
  const _Float16* W_WHH = cvtg(P_GRU_WHH, 384, 128);
  const _Float16* W_TR1 = cvtg(P_TRANS_L1_W, 128, 128);
  const _Float16* W_TR2 = cvtg(P_TRANS_L2_W, 288, 128);
  const _Float16* W_MD1 = cvtg(P_MOTDEC_L1_W, 256, 272);
  const _Float16* W_MD2 = cvtg(P_MOTDEC_L2_W, 2048, 256);
  const _Float16* W_OD1 = cvtg(P_OPADEC_L1_W, 256, 272);
  const _Float16* W_OD2 = cvtg(P_OPADEC_L2_W, 1024, 256);
  const _Float16* W_GD1 = cvtg(P_GAPDEC_L1_W, 256, 272);
  const _Float16* W_GD2 = cvtg(P_GAPDEC_L2_W, 1024, 256);
  const _Float16* W_SD1 = cvtg(P_SUNDEC_L1_W, 64, 272);
  const _Float16* W_SD2 = cvtg(P_SUNDEC_L2_W, 1, 64);
  const _Float16* W_PV1 = cvtg(P_PV_L1_W, 64, 4);
  const _Float16* W_PV2 = cvtg(P_PV_L2_W, 64, 64);
  const _Float16* W_QL1 = cvtg(P_Q_L1_W, 128, 329);
  const _Float16* W_QL2 = cvtg(P_Q_L2_W, 64, 128);
  const _Float16* W_QC  = cvtg(P_Q_CENTER_W, 1, 64);
  const _Float16* W_QLO = cvtg(P_Q_LOWER_W, 1, 64);
  const _Float16* W_QUP = cvtg(P_Q_UPPER_W, 1, 64);

  // ---- output layout (reference return order, flat f32) ----
  float* out = (float*)d_out;
  float* out_q      = out;                       // (8,3)
  float* out_trans  = out + 24;                  // (8,15,1,32,32)
  float* out_motion = out_trans + 122880;        // (8,15,2,32,32)
  float* out_opac   = out_motion + 245760;       // (8,15,1,32,32)
  float* out_gap    = out_opac + 122880;         // (8,15,1,32,32)
  float* out_sun    = out_gap + 122880;          // (8,15)
  float* out_recon  = out_sun + 120;             // (8,1,32,32)
  float* out_kl     = out_recon + 8192;          // (8,)

  // ================= encoder (all 64 frames batched) =================
  launch_conv(stream, F(IN_IMAGES), 6, 6L * 256 * 256, 0, nullptr, 0, 0, 0,
              W_ENC1, F(P_ENC_BN1_G), BN_INV, F(P_ENC_BN1_B), nullptr,
              bufA, 64, 256, 256, 2, 2, 32, 5, 5, ACT_GELU);
  launch_conv(stream, bufA, 32, 32L * 128 * 128, 0, nullptr, 0, 0, 0,
              W_RB1A, F(P_RB1_BN1_G), BN_INV, F(P_RB1_BN1_B), nullptr,
              bufB, 64, 128, 128, 1, 1, 32, 3, 3, ACT_GELU);
  launch_conv(stream, bufB, 32, 32L * 128 * 128, 0, nullptr, 0, 0, 0,
              W_RB1B, F(P_RB1_BN2_G), BN_INV, F(P_RB1_BN2_B), bufA,
              bufC, 64, 128, 128, 1, 1, 32, 3, 3, ACT_GELU);
  launch_conv(stream, bufC, 32, 32L * 128 * 128, 0, nullptr, 0, 0, 0,
              W_ENC2, F(P_ENC_BN2_G), BN_INV, F(P_ENC_BN2_B), nullptr,
              bufA, 64, 128, 128, 2, 1, 64, 3, 3, ACT_GELU);
  launch_conv(stream, bufA, 64, 64L * 64 * 64, 0, nullptr, 0, 0, 0,
              W_RB2A, F(P_RB2_BN1_G), BN_INV, F(P_RB2_BN1_B), nullptr,
              bufB, 64, 64, 64, 1, 1, 64, 3, 3, ACT_GELU);
  launch_conv(stream, bufB, 64, 64L * 64 * 64, 0, nullptr, 0, 0, 0,
              W_RB2B, F(P_RB2_BN2_G), BN_INV, F(P_RB2_BN2_B), bufA,
              bufC, 64, 64, 64, 1, 1, 64, 3, 3, ACT_GELU);
  launch_conv(stream, bufC, 64, 64L * 64 * 64, 0, nullptr, 0, 0, 0,
              W_ENC3, F(P_ENC_BN3_G), BN_INV, F(P_ENC_BN3_B), nullptr,
              bufA, 64, 64, 64, 2, 1, 96, 3, 3, ACT_GELU);
  launch_conv(stream, bufA, 96, 96L * 32 * 32, 0, nullptr, 0, 0, 0,
              W_RB3A, F(P_RB3_BN1_G), BN_INV, F(P_RB3_BN1_B), nullptr,
              bufB, 64, 32, 32, 1, 1, 96, 3, 3, ACT_GELU);
  launch_conv(stream, bufB, 96, 96L * 32 * 32, 0, nullptr, 0, 0, 0,
              W_RB3B, F(P_RB3_BN2_G), BN_INV, F(P_RB3_BN2_B), bufA,
              bufC, 64, 32, 32, 1, 1, 96, 3, 3, ACT_GELU);
  launch_conv(stream, bufC, 96, 96L * 32 * 32, 0, nullptr, 0, 0, 0,
              W_ENC4, nullptr, 1.f, F(P_ENC_B4), nullptr,
              bufA, 64, 32, 32, 1, 1, 128, 3, 3, ACT_GELU);

  // ========================= ConvLSTM (T=8) ==========================
  fill_k<<<(unsigned)((2 * 8L * 256 * 1024 + 255) / 256), 256, 0, stream>>>(
      hbuf, 0.f, 2 * 8L * 256 * 1024);
  for (int t = 0; t < 8; ++t) {
    launch_conv(stream,
                bufA, 128, 8L * 128 * 1024, (long)t * 128 * 1024,
                hbuf, 256, 256L * 1024, 0,
                W_LSTM, nullptr, 1.f, F(P_LSTM_B), nullptr,
                gates, 8, 32, 32, 1, 1, 1024, 3, 3, ACT_NONE);
    lstm_cell_k<<<(8 * 256 * 1024 + 255) / 256, 256, 0, stream>>>(gates, cbuf, hbuf, 8 * 256 * 1024);
  }

  // ===================== refine + pool + recon =======================
  launch_conv(stream, hbuf, 256, 256L * 1024, 0, nullptr, 0, 0, 0,
              W_REF1, F(P_REF_BN1_G), BN_INV, F(P_REF_BN1_B), nullptr,
              tmp1, 8, 32, 32, 1, 1, 256, 3, 3, ACT_GELU);
  launch_conv(stream, tmp1, 256, 256L * 1024, 0, nullptr, 0, 0, 0,
              W_REF2, F(P_REF_BN2_G), BN_INV, F(P_REF_BN2_B), hbuf,
              tmp2, 8, 32, 32, 1, 1, 256, 3, 3, ACT_GELU);
  mean_pool_k<<<8 * 256, 256, 0, stream>>>(tmp2, pooled, 1024);
  launch_conv(stream, tmp2, 256, 256L * 1024, 0, nullptr, 0, 0, 0,
              W_RECA, nullptr, 1.f, F(P_REC_B1), nullptr,
              tmp1, 8, 32, 32, 1, 1, 128, 3, 3, ACT_GELU);
  launch_conv(stream, tmp1, 128, 128L * 1024, 0, nullptr, 0, 0, 0,
              W_RECB, nullptr, 1.f, F(P_REC_B2), nullptr,
              out_recon, 8, 32, 32, 1, 0, 1, 1, 1, ACT_NONE);

  // ===================== latent heads + z + kl =======================
  launch_gemm(stream, pooled, 256, nullptr, 0, W_LM1, F(P_LAT_MOT_L1_B), t256, 256, 0, 256, ACT_GELU, 1.f);
  launch_gemm(stream, t256, 256, nullptr, 0, W_LM2, F(P_LAT_MOT_L2_B), st_mot, 128, 0, 128, ACT_NONE, 1.f);
  launch_gemm(stream, pooled, 256, nullptr, 0, W_LO1, F(P_LAT_OPA_L1_B), t256, 256, 0, 256, ACT_GELU, 1.f);
  launch_gemm(stream, t256, 256, nullptr, 0, W_LO2, F(P_LAT_OPA_L2_B), st_opa, 64, 0, 64, ACT_NONE, 1.f);
  launch_gemm(stream, pooled, 256, nullptr, 0, W_LG1, F(P_LAT_GAP_L1_B), t256, 256, 0, 256, ACT_GELU, 1.f);
  launch_gemm(stream, t256, 256, nullptr, 0, W_LG2, F(P_LAT_GAP_L2_B), st_gap, 64, 0, 64, ACT_NONE, 1.f);
  launch_gemm(stream, pooled, 256, nullptr, 0, W_LS1, F(P_LAT_SUN_L1_B), t256, 256, 0, 256, ACT_GELU, 1.f);
  launch_gemm(stream, t256, 256, nullptr, 0, W_LS2, F(P_LAT_SUN_L2_B), st_sun, 32, 0, 32, ACT_NONE, 1.f);
  latent_finalize_k<<<1, 256, 0, stream>>>(st_mot, st_opa, st_gap, st_sun, zbuf, out_kl);

  launch_gemm(stream, zbuf, 144, nullptr, 0, W_HID, F(P_HID_B), hid, 128, 0, 128, ACT_TANH, 1.f);
  copy_mat_k<<<(8 * 144 + 255) / 256, 256, 0, stream>>>(czbuf, 144, zbuf, 144, 8, 144);

  // ===================== GRU rollout (T_FUT=15) ======================
  for (int st = 0; st < 15; ++st) {
    launch_gemm(stream, czbuf, 144, nullptr, 0, W_WIH, F(P_GRU_BIH), gibuf, 384, 0, 384, ACT_NONE, 1.f);
    launch_gemm(stream, hid, 128, nullptr, 0, W_WHH, F(P_GRU_BHH), ghbuf, 384, 0, 384, ACT_NONE, 1.f);
    gru_cell_k<<<(8 * 128 + 255) / 256, 256, 0, stream>>>(gibuf, ghbuf, hid, 128, 8 * 128);

    launch_gemm(stream, hid, 128, nullptr, 0, W_TR1, F(P_TRANS_L1_B), t256, 128, 0, 128, ACT_GELU, 1.f);
    launch_gemm(stream, t256, 128, nullptr, 0, W_TR2, F(P_TRANS_L2_B), tsbuf, 288, 0, 288, ACT_NONE, 1.f);
    copy_mat_k<<<(8 * 144 + 255) / 256, 256, 0, stream>>>(czbuf, 144, tsbuf, 288, 8, 144);

    launch_gemm(stream, czbuf, 144, hid, 128, W_MD1, F(P_MOTDEC_L1_B), t256, 256, 0, 256, ACT_GELU, 1.f);
    launch_gemm(stream, t256, 256, nullptr, 0, W_MD2, F(P_MOTDEC_L2_B),
                out_motion, 15L * 2048, (long)st * 2048, 2048, ACT_TANH, 2.5f);

    launch_gemm(stream, czbuf, 144, hid, 128, W_OD1, F(P_OPADEC_L1_B), t256, 256, 0, 256, ACT_GELU, 1.f);
    launch_gemm(stream, t256, 256, nullptr, 0, W_OD2, F(P_OPADEC_L2_B),
                out_opac, 15L * 1024, (long)st * 1024, 1024, ACT_SIGMOID, 1.f);

    launch_gemm(stream, czbuf, 144, hid, 128, W_GD1, F(P_GAPDEC_L1_B), t256, 256, 0, 256, ACT_GELU, 1.f);
    launch_gemm(stream, t256, 256, nullptr, 0, W_GD2, F(P_GAPDEC_L2_B),
                out_gap, 15L * 1024, (long)st * 1024, 1024, ACT_SIGMOID, 1.f);

    launch_gemm(stream, czbuf, 144, hid, 128, W_SD1, F(P_SUNDEC_L1_B), t64, 64, 0, 64, ACT_GELU, 1.f);
    launch_gemm(stream, t64, 64, nullptr, 0, W_SD2, F(P_SUNDEC_L2_B),
                out_sun, 15, st, 1, ACT_SIGMOID, 1.f);
  }

  transm_k<<<(8 * 15 * 1024 + 255) / 256, 256, 0, stream>>>(out_opac, out_gap, out_trans, 8 * 15 * 1024);
  sun_attn_k<<<8, 256, 0, stream>>>(F(IN_SUNXY), out_trans, out_gap, out_opac, str_, sgap_, sop_);

  launch_gemm(stream, F(IN_PV), 4, nullptr, 0, W_PV1, F(P_PV_L1_B), pvt, 64, 0, 64, ACT_GELU, 1.f);
  launch_gemm(stream, pvt, 64, nullptr, 0, W_PV2, F(P_PV_L2_B), pvfeat, 64, 0, 64, ACT_GELU, 1.f);

  qin_build_k<<<(8 * 329 + 255) / 256, 256, 0, stream>>>(pooled, pvfeat, F(IN_SOLAR), str_, sgap_, sop_, out_sun, qinb);
  launch_gemm(stream, qinb, 329, nullptr, 0, W_QL1, F(P_Q_L1_B), f1, 128, 0, 128, ACT_GELU, 1.f);
  launch_gemm(stream, f1, 128, nullptr, 0, W_QL2, F(P_Q_L2_B), f2, 64, 0, 64, ACT_GELU, 1.f);
  launch_gemm(stream, f2, 64, nullptr, 0, W_QC, F(P_Q_CENTER_B), qc, 1, 0, 1, ACT_NONE, 1.f);
  launch_gemm(stream, f2, 64, nullptr, 0, W_QLO, F(P_Q_LOWER_B), ql, 1, 0, 1, ACT_NONE, 1.f);
  launch_gemm(stream, f2, 64, nullptr, 0, W_QUP, F(P_Q_UPPER_B), qu, 1, 0, 1, ACT_NONE, 1.f);
  quantile_k<<<1, 32, 0, stream>>>(qc, ql, qu, out_q);
}
